// Model_59090160058430
// MI455X (gfx1250) — compile-verified, matrix operand order checked by reading, not run on hardware
//
#include <hip/hip_runtime.h>
#include <hip/hip_bf16.h>

typedef _Float16 f16;
typedef _Float16 v16h __attribute__((ext_vector_type(16)));
typedef float    v8f  __attribute__((ext_vector_type(8)));

#define IN_DIM   512
#define HID      256
#define TEMP_INV 2.0f      // 1/TEMP, TEMP=0.5
#define LAM      0.001f
#define BN_EPS   1e-5f
#define NORM_EPS 1e-12f
#define JCHUNKS  4

// ---------------------------------------------------------------- utilities
__global__ void zero_kernel(float* __restrict__ p, long long n) {
    long long i = (long long)blockIdx.x * blockDim.x + threadIdx.x;
    if (i < n) p[i] = 0.f;
}

__global__ void degree_kernel(const int* __restrict__ src, const int* __restrict__ dst,
                              float* __restrict__ deg_in, float* __restrict__ deg_out, int E) {
    int e = blockIdx.x * blockDim.x + threadIdx.x;
    if (e >= E) return;
    atomicAdd(&deg_in[dst[e]], 1.f);
    atomicAdd(&deg_out[src[e]], 1.f);
}

__global__ void invsqrt_kernel(const float* __restrict__ deg_in, float* __restrict__ din_is,
                               float* __restrict__ dout_io, int N) {
    int i = blockIdx.x * blockDim.x + threadIdx.x;
    if (i >= N) return;
    float a = deg_in[i];
    din_is[i] = (a > 0.f) ? rsqrtf(a) : 0.f;
    float b = dout_io[i];
    dout_io[i] = (b > 0.f) ? rsqrtf(b) : 0.f;
}

// ------------------------------------------------------- f32 tiled GEMM
// C[M x Nc] = transformA(A[M x K]) @ W[K x Nc] (+ bias)
// 64x64 block tile, 256 threads, 4x4 register micro-tile per thread.
// AMODE: 0 = none, 1 = per-row scale (GCN out-degree), 2 = BN(train)+ReLU
template<int AMODE>
__global__ void __launch_bounds__(256)
gemm_kernel(const float* __restrict__ A, const float* __restrict__ W,
            const float* __restrict__ bias, float* __restrict__ C,
            int M, int K, int Nc,
            const float* __restrict__ rowscale,
            const float* __restrict__ mu, const float* __restrict__ rstd,
            const float* __restrict__ gamma, const float* __restrict__ beta,
            int add_bias) {
    __shared__ float As[16][65];     // [k][m]
    __shared__ float Ws[16][65];     // [k][n]
    const int t  = threadIdx.x;
    const int r0 = blockIdx.y * 64;
    const int c0 = blockIdx.x * 64;  // grid.x == Nc/64, columns always in range
    const int tx = t & 15;           // n-group (4 cols each)
    const int ty = t >> 4;           // m-group (4 rows each)
    const int a_c  = t & 15;         // k index for A loads
    const int a_r0 = t >> 4;         // A rows a_r0 + 16*i
    const int b_c  = t & 63;         // n index for W loads
    const int b_r0 = t >> 6;         // k rows b_r0 + 4*i

    float acc[4][4];
    #pragma unroll
    for (int i = 0; i < 4; ++i)
        #pragma unroll
        for (int j = 0; j < 4; ++j) acc[i][j] = 0.f;

    for (int k0 = 0; k0 < K; k0 += 16) {
        float muk = 0.f, rsk = 0.f, gk = 0.f, bk = 0.f;
        if (AMODE == 2) {
            int kk = k0 + a_c;
            muk = mu[kk]; rsk = rstd[kk]; gk = gamma[kk]; bk = beta[kk];
        }
        #pragma unroll
        for (int i = 0; i < 4; ++i) {
            int r = r0 + a_r0 + 16 * i;
            float a = 0.f;
            if (r < M) {
                a = A[(long long)r * K + (k0 + a_c)];
                if (AMODE == 1) a *= rowscale[r];
                if (AMODE == 2) {
                    float v = (a - muk) * rsk * gk + bk;
                    a = v > 0.f ? v : 0.f;
                }
            }
            As[a_c][a_r0 + 16 * i] = a;
        }
        #pragma unroll
        for (int i = 0; i < 4; ++i) {
            int kr = b_r0 + 4 * i;
            Ws[kr][b_c] = W[(long long)(k0 + kr) * Nc + (c0 + b_c)];
        }
        __syncthreads();
        #pragma unroll
        for (int kk = 0; kk < 16; ++kk) {
            float av[4], bv[4];
            #pragma unroll
            for (int i = 0; i < 4; ++i) av[i] = As[kk][ty * 4 + i];
            #pragma unroll
            for (int j = 0; j < 4; ++j) bv[j] = Ws[kk][tx * 4 + j];
            #pragma unroll
            for (int i = 0; i < 4; ++i)
                #pragma unroll
                for (int j = 0; j < 4; ++j) acc[i][j] += av[i] * bv[j];
        }
        __syncthreads();
    }
    #pragma unroll
    for (int i = 0; i < 4; ++i) {
        int r = r0 + ty * 4 + i;
        if (r < M) {
            #pragma unroll
            for (int j = 0; j < 4; ++j) {
                int c = c0 + tx * 4 + j;
                float v = acc[i][j];
                if (add_bias) v += bias[c];
                C[(long long)r * Nc + c] = v;
            }
        }
    }
}

// ----------------------------------------- GCN scatter-sum over edges (H==256)
__global__ void edge_agg_kernel(const int* __restrict__ src, const int* __restrict__ dst,
                                const float* __restrict__ H, float* __restrict__ agg,
                                long long EH) {
    long long t = (long long)blockIdx.x * blockDim.x + threadIdx.x;
    if (t >= EH) return;
    int e = (int)(t >> 8);
    int c = (int)(t & 255);
    atomicAdd(&agg[((long long)dst[e] << 8) + c], H[((long long)src[e] << 8) + c]);
}

__global__ void gcn_post_kernel(const float* __restrict__ agg, const float* __restrict__ din_is,
                                const float* __restrict__ b, float* __restrict__ out,
                                long long NH, int do_relu) {
    long long t = (long long)blockIdx.x * blockDim.x + threadIdx.x;
    if (t >= NH) return;
    int i = (int)(t >> 8);
    int c = (int)(t & 255);
    float v = agg[t] * din_is[i] + b[c];
    if (do_relu) v = v > 0.f ? v : 0.f;
    out[t] = v;
}

// -------------------------------------------------- BatchNorm training stats
__global__ void bn_stats_kernel(const float* __restrict__ T, int N,
                                float* __restrict__ mu, float* __restrict__ rstd) {
    __shared__ float s1[256], s2[256];
    int c = blockIdx.x;
    float sum = 0.f, sq = 0.f;
    for (int i = threadIdx.x; i < N; i += blockDim.x) {
        float v = T[((long long)i << 8) + c];
        sum += v; sq += v * v;
    }
    s1[threadIdx.x] = sum; s2[threadIdx.x] = sq;
    __syncthreads();
    for (int s = 128; s > 0; s >>= 1) {
        if ((int)threadIdx.x < s) { s1[threadIdx.x] += s1[threadIdx.x + s];
                                    s2[threadIdx.x] += s2[threadIdx.x + s]; }
        __syncthreads();
    }
    if (threadIdx.x == 0) {
        float m = s1[0] / (float)N;
        float var = s2[0] / (float)N - m * m;
        mu[c] = m;
        rstd[c] = rsqrtf(var + BN_EPS);
    }
}

// --------------------------------- L2 normalize rows (256) + f16 shadow copy
__global__ void l2norm_f16_kernel(const float* __restrict__ X, float* __restrict__ Xn,
                                  f16* __restrict__ Xh, int N) {
    int gt = blockIdx.x * blockDim.x + threadIdx.x;
    int row = gt >> 5, lane = gt & 31;
    if (row >= N) return;                      // wave-uniform
    const float* xr = X + ((long long)row << 8);
    float v[8]; float s = 0.f;
    #pragma unroll
    for (int j = 0; j < 8; ++j) { v[j] = xr[lane + 32 * j]; s += v[j] * v[j]; }
    #pragma unroll
    for (int m = 1; m < 32; m <<= 1) s += __shfl_xor(s, m, 32);
    float inv = 1.f / fmaxf(sqrtf(s), NORM_EPS);
    float* yr = Xn + ((long long)row << 8);
    f16*   hr = Xh + ((long long)row << 8);
    #pragma unroll
    for (int j = 0; j < 8; ++j) {
        float w = v[j] * inv;
        yr[lane + 32 * j] = w;
        hr[lane + 32 * j] = (f16)w;
    }
}

// ------------------------------------- pos score: wave per edge, 256-dim dot
__global__ void pos_kernel(const int* __restrict__ src, const int* __restrict__ dst,
                           const float* __restrict__ z, const float* __restrict__ q,
                           float* __restrict__ posacc, int E) {
    int gt = blockIdx.x * blockDim.x + threadIdx.x;
    int e = gt >> 5, lane = gt & 31;
    if (e >= E) return;                        // wave-uniform
    const float* zr = z + ((long long)src[e] << 8);
    const float* qr = q + ((long long)dst[e] << 8);
    float s = 0.f;
    #pragma unroll
    for (int j = 0; j < 8; ++j) s += zr[lane + 32 * j] * qr[lane + 32 * j];
    #pragma unroll
    for (int m = 1; m < 32; m <<= 1) s += __shfl_xor(s, m, 32);
    if (lane == 0) atomicAdd(&posacc[dst[e]], s * TEMP_INV);
}

// ---------------- neg score: row sums of exp(z@z^T * 2) and exp(z@q^T * 2)
// One wave per 16-row strip x 1/JCHUNKS of the column range (blockIdx.y).
// A (16x256 f16) lives in 8 v16h regs; B tiles are double-buffered in LDS and
// filled with global_load_async_to_lds_b128 (ASYNCcnt), consumed by
// v_wmma_f32_16x16x32_f16 chains. Requires N % 16 == 0 (N = 10000 here).

// fill one 16x256-half tile (8 KB): 4 async b128 issues/thread, 16 B each lane
#define ASYNC_FILL(BASE, TILE, BUFOFF)                                           \
    {                                                                            \
        const f16* _gp = (BASE) + ((long long)(TILE) << 12);                     \
        _Pragma("unroll")                                                        \
        for (int _k = 0; _k < 4; ++_k) {                                         \
            unsigned _lds = ldsBase + (BUFOFF) + loff[_k];                       \
            unsigned long long _ga = (unsigned long long)(size_t)(_gp + goff[_k]); \
            asm volatile("global_load_async_to_lds_b128 %0, %1, off"             \
                         :: "v"(_lds), "v"(_ga) : "memory");                     \
        }                                                                        \
    }

#define NEG_PASS(BASE, PS)                                                       \
    if (t1 > t0) {                                                               \
        ASYNC_FILL(BASE, t0, 0u);                                                \
        for (int _t = t0; _t < t1; ++_t) {                                       \
            unsigned _cur = (unsigned)((_t - t0) & 1);                           \
            if (_t + 1 < t1) {                                                   \
                ASYNC_FILL(BASE, _t + 1, (_cur ^ 1u) * 8704u);                   \
                asm volatile("s_wait_asynccnt 0x4" ::: "memory");                \
            } else {                                                             \
                asm volatile("s_wait_asynccnt 0x0" ::: "memory");                \
            }                                                                    \
            __syncthreads();                                                     \
            v8f _acc = {};                                                       \
            _Pragma("unroll")                                                    \
            for (int _c = 0; _c < 8; ++_c) {                                     \
                v16h _b = *(const v16h*)(&Bs[_cur][rB][_c * 32 + koff]);         \
                _acc = __builtin_amdgcn_wmma_f32_16x16x32_f16(                   \
                           false, A[_c], false, _b, (short)0, _acc, false, false); \
            }                                                                    \
            _Pragma("unroll")                                                    \
            for (int _r = 0; _r < 8; ++_r) PS[_r] += __expf(_acc[_r] * TEMP_INV); \
            __syncthreads();                                                     \
        }                                                                        \
    }

__global__ void __launch_bounds__(128)
neg_wmma_kernel(const f16* __restrict__ zh, const f16* __restrict__ qh,
                float* __restrict__ neg_sim, float* __restrict__ neg_sim2, int N) {
    __shared__ __align__(32) f16 Bs[2][16][272];   // 272 = 256 + 16 halves pad; 8704 B/buf
    const int lane = threadIdx.x & 31;
    const int wv   = threadIdx.x >> 5;             // 0..3
    const int nstrips = N >> 4;
    int strip = blockIdx.x * 4 + wv;
    const bool valid = (strip < nstrips);
    if (!valid) strip = nstrips - 1;               // duplicate work, atomics skipped
    const int rowA = (strip << 4) + (lane & 15);
    const int koff = (lane & 16);                  // K-half select: 0 or 16
    const int rB   = lane & 15;

    v16h A[8];
    #pragma unroll
    for (int c = 0; c < 8; ++c)
        A[c] = *(const v16h*)(zh + ((long long)rowA << 8) + c * 32 + koff);

    // column-tile range for this chunk (uniform per block)
    const int per = (nstrips + JCHUNKS - 1) / JCHUNKS;
    const int t0  = blockIdx.y * per;
    int t1 = t0 + per; if (t1 > nstrips) t1 = nstrips;

    // per-thread fill addressing: 4 chunks of 16 B, full 16x512B tile coverage
    unsigned ldsBase = (unsigned)(size_t)(&Bs[0][0][0]);
    unsigned loff[4]; long long goff[4];
    #pragma unroll
    for (int k = 0; k < 4; ++k) {
        int id = (int)threadIdx.x + (k << 7);      // 0..511
        int r = id >> 5, c = id & 31;
        loff[k] = (unsigned)(r * 544 + c * 16);    // bytes in padded LDS tile
        goff[k] = ((long long)r << 8) + (c << 3);  // halves in global tile
    }

    float ps1[8], ps2[8];
    #pragma unroll
    for (int r = 0; r < 8; ++r) { ps1[r] = 0.f; ps2[r] = 0.f; }

    NEG_PASS(zh, ps1)     // sum_j exp(2 * z_i . z_j)
    NEG_PASS(qh, ps2)     // sum_j exp(2 * z_i . q_j)

    // C layout: lanes 0-15, VGPR r -> (M=r, N=lane); lanes 16-31 -> (M=r+8, N=lane-16)
    const int i0 = strip << 4;
    #pragma unroll
    for (int r = 0; r < 8; ++r) {
        float v1 = ps1[r], v2 = ps2[r];
        #pragma unroll
        for (int m = 1; m < 16; m <<= 1) {         // reduce within each 16-lane half
            v1 += __shfl_xor(v1, m, 32);
            v2 += __shfl_xor(v2, m, 32);
        }
        if (valid && lane == 0) {
            atomicAdd(&neg_sim[i0 + r], v1);
            atomicAdd(&neg_sim2[i0 + r], LAM * v2);
        }
        if (valid && lane == 16) {
            atomicAdd(&neg_sim[i0 + 8 + r], v1);
            atomicAdd(&neg_sim2[i0 + 8 + r], LAM * v2);
        }
    }
}

__global__ void neg_msg_kernel(const int* __restrict__ src, const int* __restrict__ dst,
                               const float* __restrict__ ns, const float* __restrict__ ns2,
                               float* __restrict__ negacc, int E) {
    int e = blockIdx.x * blockDim.x + threadIdx.x;
    if (e >= E) return;
    float m = logf(ns[dst[e]] + ns2[src[e]]);
    atomicAdd(&negacc[dst[e]], m);
}

__global__ void loss_kernel(const float* __restrict__ posacc, const float* __restrict__ negacc,
                            const float* __restrict__ deg_in, float* __restrict__ out, int N) {
    __shared__ float sh[256];
    int i = blockIdx.x * blockDim.x + threadIdx.x;
    float v = 0.f;
    if (i < N) v = (negacc[i] - posacc[i]) / deg_in[i];
    sh[threadIdx.x] = v;
    __syncthreads();
    for (int s = 128; s > 0; s >>= 1) {
        if ((int)threadIdx.x < s) sh[threadIdx.x] += sh[threadIdx.x + s];
        __syncthreads();
    }
    if (threadIdx.x == 0) atomicAdd(out, sh[0] / (float)N);
}

// ---------------------------------------------------------------- launcher
extern "C" void kernel_launch(void* const* d_in, const int* in_sizes, int n_in,
                              void* d_out, int out_size, void* d_ws, size_t ws_size,
                              hipStream_t stream) {
    const float* feat  = (const float*)d_in[0];
    const int*   src   = (const int*)d_in[1];
    const int*   dst   = (const int*)d_in[2];
    const float* W1    = (const float*)d_in[3];
    const float* b1    = (const float*)d_in[4];
    const float* W2    = (const float*)d_in[5];
    const float* b2    = (const float*)d_in[6];
    const float* Wt1   = (const float*)d_in[7];
    const float* bt1   = (const float*)d_in[8];
    const float* gamma = (const float*)d_in[9];
    const float* beta  = (const float*)d_in[10];
    const float* Wt2   = (const float*)d_in[11];
    const float* bt2   = (const float*)d_in[12];
    const float* Wp    = (const float*)d_in[13];
    const float* bp    = (const float*)d_in[14];
    float* out = (float*)d_out;
    (void)n_in; (void)out_size; (void)ws_size;

    const int N = in_sizes[0] / IN_DIM;
    const int E = in_sizes[1];
    const long long NH = (long long)N * HID;

    size_t off = 0;
    auto alloc = [&](size_t bytes) -> char* {
        off = (off + 255) & ~(size_t)255;
        char* p = (char*)d_ws + off;
        off += bytes;
        return p;
    };
    float* deg_in  = (float*)alloc((size_t)N * 4);
    float* din_is  = (float*)alloc((size_t)N * 4);
    float* dout_is = (float*)alloc((size_t)N * 4);
    float* posacc  = (float*)alloc((size_t)N * 4);
    float* negacc  = (float*)alloc((size_t)N * 4);
    float* ns      = (float*)alloc((size_t)N * 4);
    float* ns2     = (float*)alloc((size_t)N * 4);
    float* mu      = (float*)alloc((size_t)HID * 4);
    float* rstd    = (float*)alloc((size_t)HID * 4);
    float* tmp     = (float*)alloc((size_t)NH * 4);
    float* agg     = (float*)alloc((size_t)NH * 4);
    float* h1      = (float*)alloc((size_t)NH * 4);
    float* hfin    = (float*)alloc((size_t)NH * 4);
    f16*   zh      = (f16*)alloc((size_t)NH * 2);
    f16*   qh      = (f16*)alloc((size_t)NH * 2);
    float* tbuf  = h1;    // reuse after GCN consumed h1
    float* trans = tmp;   // reuse after GCN layer-2 aggregation done
    float* qbuf  = agg;   // reuse after GCN layer-2 post

    dim3 b256(256);
    auto g1 = [&](long long n) { return dim3((unsigned)((n + 255) / 256)); };

    // zero accumulators (workspace is poisoned before timing)
    zero_kernel<<<g1(N), b256, 0, stream>>>(deg_in, N);
    zero_kernel<<<g1(N), b256, 0, stream>>>(dout_is, N);
    zero_kernel<<<g1(N), b256, 0, stream>>>(posacc, N);
    zero_kernel<<<g1(N), b256, 0, stream>>>(negacc, N);
    zero_kernel<<<g1(N), b256, 0, stream>>>(ns, N);
    zero_kernel<<<g1(N), b256, 0, stream>>>(ns2, N);
    zero_kernel<<<1, 32, 0, stream>>>(out, 1);

    degree_kernel<<<g1(E), b256, 0, stream>>>(src, dst, deg_in, dout_is, E);
    invsqrt_kernel<<<g1(N), b256, 0, stream>>>(deg_in, din_is, dout_is, N);

    dim3 gemmB(256);
    dim3 gemmG(HID / 64, (unsigned)((N + 63) / 64));

    // --- GCN layer 1 ---
    gemm_kernel<1><<<gemmG, gemmB, 0, stream>>>(feat, W1, nullptr, tmp, N, IN_DIM, HID,
                                                dout_is, nullptr, nullptr, nullptr, nullptr, 0);
    zero_kernel<<<g1(NH), b256, 0, stream>>>(agg, NH);
    edge_agg_kernel<<<g1((long long)E * HID), b256, 0, stream>>>(src, dst, tmp, agg,
                                                                 (long long)E * HID);
    gcn_post_kernel<<<g1(NH), b256, 0, stream>>>(agg, din_is, b1, h1, NH, 1);

    // --- GCN layer 2 ---
    gemm_kernel<1><<<gemmG, gemmB, 0, stream>>>(h1, W2, nullptr, tmp, N, HID, HID,
                                                dout_is, nullptr, nullptr, nullptr, nullptr, 0);
    zero_kernel<<<g1(NH), b256, 0, stream>>>(agg, NH);
    edge_agg_kernel<<<g1((long long)E * HID), b256, 0, stream>>>(src, dst, tmp, agg,
                                                                 (long long)E * HID);
    gcn_post_kernel<<<g1(NH), b256, 0, stream>>>(agg, din_is, b2, hfin, NH, 0);

    // --- target MLP: Linear -> BN(train) -> ReLU -> Linear ---
    gemm_kernel<0><<<gemmG, gemmB, 0, stream>>>(feat, Wt1, bt1, tbuf, N, IN_DIM, HID,
                                                nullptr, nullptr, nullptr, nullptr, nullptr, 1);
    bn_stats_kernel<<<HID, b256, 0, stream>>>(tbuf, N, mu, rstd);
    gemm_kernel<2><<<gemmG, gemmB, 0, stream>>>(tbuf, Wt2, bt2, trans, N, HID, HID,
                                                nullptr, mu, rstd, gamma, beta, 1);

    // --- projector ---
    gemm_kernel<0><<<gemmG, gemmB, 0, stream>>>(trans, Wp, bp, qbuf, N, HID, HID,
                                                nullptr, nullptr, nullptr, nullptr, nullptr, 1);

    // --- normalize + f16 shadow ---
    l2norm_f16_kernel<<<g1((long long)N * 32), b256, 0, stream>>>(hfin, hfin, zh, N);
    l2norm_f16_kernel<<<g1((long long)N * 32), b256, 0, stream>>>(qbuf, qbuf, qh, N);

    // --- pos score ---
    pos_kernel<<<g1((long long)E * 32), b256, 0, stream>>>(src, dst, hfin, qbuf, posacc, E);

    // --- neg score (WMMA N x N, async-to-LDS double buffered, j split 4 ways) ---
    int nstrips = N / 16;
    neg_wmma_kernel<<<dim3((nstrips + 3) / 4, JCHUNKS), dim3(128), 0, stream>>>(zh, qh, ns, ns2, N);

    // --- edge log message + final mean ---
    neg_msg_kernel<<<g1(E), b256, 0, stream>>>(src, dst, ns, ns2, negacc, E);
    loss_kernel<<<g1(N), b256, 0, stream>>>(posacc, negacc, deg_in, out, N);
}
